// MSA_head_25701084299627
// MI455X (gfx1250) — matrix operand order repaired and checked by reading, not compiled, not run on hardware
//
#include <hip/hip_runtime.h>
#include <hip/hip_bf16.h>
#include <math.h>

typedef __attribute__((ext_vector_type(16))) _Float16 v16h;
typedef __attribute__((ext_vector_type(8)))  float    v8f;
typedef __attribute__((ext_vector_type(2)))  float    v2f;

#define HWC   4096      // H*W
#define CC    128       // channels
#define BB    8         // batch
#define HEADS 8
#define CHD   16        // channels per head
#define HIDN  512

// ---------------- channel LayerNorm (per pixel over 128 channels) ------------
__global__ void ln_kernel(const float* __restrict__ X, const float* __restrict__ w,
                          const float* __restrict__ bias, float* __restrict__ Y) {
    int t = blockIdx.x * blockDim.x + threadIdx.x;     // 0 .. B*HW-1
    int b = t >> 12;                                   // HW = 4096
    int p = t & 4095;
    const float* xp = X + ((size_t)b * CC) * HWC + p;
    float s = 0.f, s2 = 0.f;
    #pragma unroll 4
    for (int c = 0; c < CC; ++c) { float v = xp[(size_t)c * HWC]; s += v; s2 += v * v; }
    float mu  = s * (1.0f / CC);
    float var = s2 * (1.0f / CC) - mu * mu;
    float inv = rsqrtf(var + 1e-5f);
    float* yp = Y + ((size_t)b * CC) * HWC + p;
    #pragma unroll 4
    for (int c = 0; c < CC; ++c)
        yp[(size_t)c * HWC] = (xp[(size_t)c * HWC] - mu) * inv * w[c] + bias[c];
}

// ---------------- 1x1 conv == GEMM via v_wmma_f32_16x16x32_f16 ---------------
// O[b, och_off+m, n] = sum_c Wm[m,c] * X[b,c,n]  (+ residual R[b,m,n] if HASR)
// grid: (HW/64, Cout/64, B), block: 128 (4 waves); each wave: 64(M) x 16(N)
template<bool HASR>
__global__ void conv1x1_wmma(const float* __restrict__ X, const float* __restrict__ Wm,
                             const float* __restrict__ R, float* __restrict__ O,
                             int Cin, int Cout, int ostride_ch, int och_off) {
    const int wid  = threadIdx.x >> 5;                  // wave in block
    const int l    = threadIdx.x & 31;
    const int half = l >> 4;
    const int lm   = l & 15;
    const int n0   = ((blockIdx.x << 2) + wid) << 4;    // pixel tile
    const int m0   = blockIdx.y << 6;                   // 64 output channels
    const int b    = blockIdx.z;

    const float* Xb = X + (size_t)b * Cin * HWC + n0 + lm;   // this lane's column
    const float* Xpf = X + (size_t)b * Cin * HWC + n0;       // prefetch base
    v8f acc[4] = {};

    for (int k0 = 0; k0 < Cin; k0 += 32) {
        // ---- B tile (activations) 32x16 f16: lane col n=lm, element e -> K=half*8+e (e<8), 16+half*8+(e-8) (e>=8)
        v16h bt;
        const float* xk = Xb + (size_t)(k0 + (half << 3)) * HWC;
        #pragma unroll
        for (int e = 0; e < 16; ++e) bt[e] = (_Float16)xk[(size_t)(e < 8 ? e : e + 8) * HWC];

        // ---- prefetch next B block: lane l covers K-row k0+32+l (global_prefetch_b8)
        if (k0 + 32 < Cin)
            __builtin_prefetch(Xpf + (size_t)(k0 + 32 + l) * HWC, 0, 1);

        // ---- 4 A sub-tiles (weights), vector b128 loads; per-lane K runs are
        //      [half*8, half*8+8) and [16+half*8, 16+half*8+8)
        #pragma unroll
        for (int t = 0; t < 4; ++t) {
            const float* wr = Wm + (size_t)(m0 + (t << 4) + lm) * Cin + k0 + (half << 3);
            float4 w0 = *(const float4*)(wr);
            float4 w1 = *(const float4*)(wr + 4);
            float4 w2 = *(const float4*)(wr + 16);
            float4 w3 = *(const float4*)(wr + 20);
            v16h at;
            at[0]  = (_Float16)w0.x; at[1]  = (_Float16)w0.y;
            at[2]  = (_Float16)w0.z; at[3]  = (_Float16)w0.w;
            at[4]  = (_Float16)w1.x; at[5]  = (_Float16)w1.y;
            at[6]  = (_Float16)w1.z; at[7]  = (_Float16)w1.w;
            at[8]  = (_Float16)w2.x; at[9]  = (_Float16)w2.y;
            at[10] = (_Float16)w2.z; at[11] = (_Float16)w2.w;
            at[12] = (_Float16)w3.x; at[13] = (_Float16)w3.y;
            at[14] = (_Float16)w3.z; at[15] = (_Float16)w3.w;
            acc[t] = __builtin_amdgcn_wmma_f32_16x16x32_f16(false, at, false, bt,
                                                            (short)0, acc[t], false, false);
        }
    }

    // ---- epilogue: D vgpr v -> row m = v + 8*half, col = lm
    #pragma unroll
    for (int t = 0; t < 4; ++t) {
        const int ocb = m0 + (t << 4) + (half << 3);
        float* ob = O + ((size_t)b * ostride_ch + och_off + ocb) * HWC + n0 + lm;
        const float* rb = HASR ? (R + ((size_t)b * Cout + ocb) * HWC + n0 + lm) : nullptr;
        #pragma unroll
        for (int vv = 0; vv < 8; ++vv) {
            float val = acc[t][vv];
            if (HASR) val += rb[(size_t)vv * HWC];
            ob[(size_t)vv * HWC] = val;
        }
    }
}

// ---------------- depthwise conv (same padding) ------------------------------
// grid: (HW/256, Cn, B), block 256
__global__ void dwconv_kernel(const float* __restrict__ X, const float* __restrict__ Wk,
                              float* __restrict__ Y, int Cn, int ks) {
    int p = (blockIdx.x << 8) + threadIdx.x;
    int yy = p >> 6, xx = p & 63;
    int c = blockIdx.y, b = blockIdx.z;
    const float* xp = X + ((size_t)b * Cn + c) * HWC;
    const float* wp = Wk + (size_t)c * ks * ks;
    int r = ks >> 1;
    float s = 0.f;
    for (int i = 0; i < ks; ++i) {
        int y2 = yy + i - r; if (y2 < 0 || y2 > 63) continue;
        for (int j = 0; j < ks; ++j) {
            int x2 = xx + j - r; if (x2 < 0 || x2 > 63) continue;
            s += xp[(y2 << 6) + x2] * wp[i * ks + j];
        }
    }
    Y[((size_t)b * Cn + c) * HWC + p] = s;
}

// ------------- depthwise conv fused with GEGLU (FFN branch) ------------------
// X: (B, 2*HID, HW). channel c -> gate, c+HID -> mult.  G[b, off+c, p] = gelu(g)*m
__global__ void dwconv_geglu_kernel(const float* __restrict__ X, const float* __restrict__ Wk,
                                    float* __restrict__ G, int ks, int och_off) {
    int p = (blockIdx.x << 8) + threadIdx.x;
    int yy = p >> 6, xx = p & 63;
    int c = blockIdx.y, b = blockIdx.z;
    int r = ks >> 1;
    const float* xg = X + ((size_t)b * (2 * HIDN) + c) * HWC;
    const float* xm = X + ((size_t)b * (2 * HIDN) + c + HIDN) * HWC;
    const float* wg = Wk + (size_t)c * ks * ks;
    const float* wm = Wk + (size_t)(c + HIDN) * ks * ks;
    float g = 0.f, m = 0.f;
    for (int i = 0; i < ks; ++i) {
        int y2 = yy + i - r; if (y2 < 0 || y2 > 63) continue;
        for (int j = 0; j < ks; ++j) {
            int x2 = xx + j - r; if (x2 < 0 || x2 > 63) continue;
            float wgt_g = wg[i * ks + j], wgt_m = wm[i * ks + j];
            float vg = xg[(y2 << 6) + x2], vm = xm[(y2 << 6) + x2];
            g += vg * wgt_g; m += vm * wgt_m;
        }
    }
    float gelu = 0.5f * g * (1.0f + erff(g * 0.70710678118654752f));
    G[((size_t)b * (3 * HIDN) + och_off + c) * HWC + p] = gelu * m;
}

// ---------------- in-place L2 row normalization (rows of 4096) ---------------
__global__ void rownorm_kernel(float* __restrict__ X) {
    float* row = X + (size_t)blockIdx.x * HWC;
    __shared__ float red[8];
    float s = 0.f;
    for (int i = threadIdx.x; i < HWC; i += 256) { float v = row[i]; s += v * v; }
    #pragma unroll
    for (int off = 16; off > 0; off >>= 1) s += __shfl_down(s, off, 32);
    if ((threadIdx.x & 31) == 0) red[threadIdx.x >> 5] = s;
    __syncthreads();
    if (threadIdx.x == 0) {
        float t = 0.f;
        #pragma unroll
        for (int i = 0; i < 8; ++i) t += red[i];
        red[0] = 1.0f / fmaxf(sqrtf(t), 1e-12f);
    }
    __syncthreads();
    float inv = red[0];
    for (int i = threadIdx.x; i < HWC; i += 256) row[i] *= inv;
}

// ----- channel attention per (b,h): S = Qn Kn^T (K=4096, f32 WMMA), ----------
// ----- softmax(S*temp), O = S V (K=16).  One wave per (b,h). ----------------
__global__ void attn_kernel(const float* __restrict__ Q, const float* __restrict__ K,
                            const float* __restrict__ V, const float* __restrict__ temp,
                            float* __restrict__ O, int och_off) {
    int h = blockIdx.x, b = blockIdx.y;
    int l = threadIdx.x, half = l >> 4, lm = l & 15;
    const float* q = Q + ((size_t)(b * HEADS + h) * CHD) * HWC;
    const float* k = K + ((size_t)(b * HEADS + h) * CHD) * HWC;
    const float* v = V + ((size_t)(b * HEADS + h) * CHD) * HWC;

    // ---- scores: 16x16 output, K-reduction over 4096 via v_wmma_f32_16x16x4_f32
    v8f s = {};
    const float* qr = q + (size_t)lm * HWC + 2 * half;  // A row m = lm
    const float* kr = k + (size_t)lm * HWC + 2 * half;  // B col d = lm
    for (int k0 = 0; k0 < HWC; k0 += 4) {
        v2f a, bm;
        a[0]  = qr[k0];  a[1]  = qr[k0 + 1];
        bm[0] = kr[k0];  bm[1] = kr[k0 + 1];
        s = __builtin_amdgcn_wmma_f32_16x16x4_f32(false, a, false, bm,
                                                  (short)0, s, false, false);
    }

    // ---- softmax over d (columns), temperature scale, via LDS
    __shared__ float sm[16][17];
    #pragma unroll
    for (int vv = 0; vv < 8; ++vv) sm[vv + (half << 3)][lm] = s[vv];
    __syncthreads();
    float t = temp[h];
    if (l < 16) {
        float mx = -3.4e38f;
        #pragma unroll
        for (int d = 0; d < 16; ++d) { float z = sm[l][d] * t; sm[l][d] = z; mx = fmaxf(mx, z); }
        float sum = 0.f;
        #pragma unroll
        for (int d = 0; d < 16; ++d) { float e = __expf(sm[l][d] - mx); sm[l][d] = e; sum += e; }
        float inv = 1.0f / sum;
        #pragma unroll
        for (int d = 0; d < 16; ++d) sm[l][d] *= inv;
    }
    __syncthreads();

    // ---- apply: O[c, n] = sum_d attn[c,d] * v[d,n], 4 f32 WMMAs per 16-pixel tile
    float* o = O + ((size_t)b * (3 * CC) + och_off + h * CHD) * HWC;
    for (int n0 = 0; n0 < HWC; n0 += 16) {
        v8f acc = {};
        #pragma unroll
        for (int k0 = 0; k0 < 16; k0 += 4) {
            v2f a, bm;
            a[0]  = sm[lm][k0 + 2 * half];
            a[1]  = sm[lm][k0 + 2 * half + 1];
            bm[0] = v[(size_t)(k0 + 2 * half) * HWC + n0 + lm];
            bm[1] = v[(size_t)(k0 + 2 * half + 1) * HWC + n0 + lm];
            acc = __builtin_amdgcn_wmma_f32_16x16x4_f32(false, a, false, bm,
                                                        (short)0, acc, false, false);
        }
        #pragma unroll
        for (int vv = 0; vv < 8; ++vv)
            o[(size_t)(vv + (half << 3)) * HWC + n0 + lm] = acc[vv];
    }
}

extern "C" void kernel_launch(void* const* d_in, const int* in_sizes, int n_in,
                              void* d_out, int out_size, void* d_ws, size_t ws_size,
                              hipStream_t stream) {
    (void)in_sizes; (void)n_in; (void)out_size; (void)ws_size;
    const float* x      = (const float*)d_in[0];
    const float* ln1_w  = (const float*)d_in[1];
    const float* ln1_b  = (const float*)d_in[2];
    const float* temp   = (const float*)d_in[3];
    const float* wq     = (const float*)d_in[4];
    const float* wk     = (const float*)d_in[5];
    const float* wv     = (const float*)d_in[6];
    const float* dwq[3] = { (const float*)d_in[7],  (const float*)d_in[10], (const float*)d_in[13] };
    const float* dwk[3] = { (const float*)d_in[8],  (const float*)d_in[11], (const float*)d_in[14] };
    const float* dwv[3] = { (const float*)d_in[9],  (const float*)d_in[12], (const float*)d_in[15] };
    const float* attn_po= (const float*)d_in[16];
    const float* ln2_w  = (const float*)d_in[17];
    const float* ln2_b  = (const float*)d_in[18];
    const float* ffn_in = (const float*)d_in[19];
    const float* fdw[3] = { (const float*)d_in[20], (const float*)d_in[21], (const float*)d_in[22] };
    const float* ffn_po = (const float*)d_in[23];
    float* out = (float*)d_out;

    float* ws = (float*)d_ws;
    const size_t szC = (size_t)BB * CC * HWC;                 // 4.19M floats
    float* xn   = ws;                                         // B*128*HW (reused for ln2)
    float* x1   = xn  + szC;                                  // B*128*HW
    float* ocat = x1  + szC;                                  // B*384*HW
    float* q0   = ocat + (size_t)BB * 3 * CC * HWC;
    float* k0b  = q0  + szC;
    float* v0   = k0b + szC;
    float* qd   = v0  + szC;
    float* kd   = qd  + szC;
    float* vd   = kd  + szC;
    float* xin  = q0;                                         // reuse (B*1024*HW), q/k/v dead
    float* gcat = xin + (size_t)BB * 2 * HIDN * HWC;          // B*1536*HW

    dim3 blk128(128);
    dim3 gGemmC(HWC / 64, CC / 64, BB);                       // Cout = 128

    // ---- attention branch ----
    ln_kernel<<<BB * HWC / 256, 256, 0, stream>>>(x, ln1_w, ln1_b, xn);
    conv1x1_wmma<false><<<gGemmC, blk128, 0, stream>>>(xn, wq, nullptr, q0,  CC, CC, CC, 0);
    conv1x1_wmma<false><<<gGemmC, blk128, 0, stream>>>(xn, wk, nullptr, k0b, CC, CC, CC, 0);
    conv1x1_wmma<false><<<gGemmC, blk128, 0, stream>>>(xn, wv, nullptr, v0,  CC, CC, CC, 0);

    dim3 gDw(HWC / 256, CC, BB);
    for (int sidx = 0; sidx < 3; ++sidx) {
        int ks = 3 + 2 * sidx;
        dwconv_kernel<<<gDw, 256, 0, stream>>>(q0,  dwq[sidx], qd, CC, ks);
        dwconv_kernel<<<gDw, 256, 0, stream>>>(k0b, dwk[sidx], kd, CC, ks);
        dwconv_kernel<<<gDw, 256, 0, stream>>>(v0,  dwv[sidx], vd, CC, ks);
        rownorm_kernel<<<BB * CC, 256, 0, stream>>>(qd);
        rownorm_kernel<<<BB * CC, 256, 0, stream>>>(kd);
        attn_kernel<<<dim3(HEADS, BB), dim3(32), 0, stream>>>(qd, kd, vd, temp, ocat, sidx * CC);
    }
    // x1 = x + attn_po @ ocat
    conv1x1_wmma<true><<<gGemmC, blk128, 0, stream>>>(ocat, attn_po, x, x1, 3 * CC, CC, CC, 0);

    // ---- FFN branch ----
    ln_kernel<<<BB * HWC / 256, 256, 0, stream>>>(x1, ln2_w, ln2_b, xn);
    conv1x1_wmma<false><<<dim3(HWC / 64, 2 * HIDN / 64, BB), blk128, 0, stream>>>(
        xn, ffn_in, nullptr, xin, CC, 2 * HIDN, 2 * HIDN, 0);
    dim3 gDwF(HWC / 256, HIDN, BB);
    for (int sidx = 0; sidx < 3; ++sidx) {
        int ks = 3 + 2 * sidx;
        dwconv_geglu_kernel<<<gDwF, 256, 0, stream>>>(xin, fdw[sidx], gcat, ks, sidx * HIDN);
    }
    // out = x1 + ffn_po @ gcat
    conv1x1_wmma<true><<<gGemmC, blk128, 0, stream>>>(gcat, ffn_po, x1, out, 3 * HIDN, CC, CC, 0);
}
